// MultiHeadLatentAttention_31516470018397
// MI455X (gfx1250) — matrix operand order repaired and checked by reading, not compiled, not run on hardware
//
#include <hip/hip_runtime.h>
#include <hip/hip_bf16.h>

// ---------------------------------------------------------------------------
// MLA attention for MI455X (gfx1250, wave32, WMMA bf16 16x16x32, f32 accum)
// GEMM tiles staged through LDS by the Tensor Data Mover (TDM) when available.
// ---------------------------------------------------------------------------

typedef __attribute__((ext_vector_type(16))) __bf16    v16bf;
typedef __attribute__((ext_vector_type(8)))  __bf16    v8bf;
typedef __attribute__((ext_vector_type(8)))  float     v8f;
typedef __attribute__((ext_vector_type(4)))  int       v4i;
typedef __attribute__((ext_vector_type(8)))  int       v8i;
typedef __attribute__((ext_vector_type(4)))  unsigned  u4i;

#if defined(__has_builtin)
#if __has_builtin(__builtin_amdgcn_tensor_load_to_lds)
#define HAVE_TDM 1
#endif
#endif

static __device__ __forceinline__ v8f zero8() {
  v8f z;
#pragma unroll
  for (int e = 0; e < 8; ++e) z[e] = 0.0f;
  return z;
}

static __device__ __forceinline__ v8f wmma_bf16(v16bf a, v16bf b, v8f c) {
  return __builtin_amdgcn_wmma_f32_16x16x32_bf16(false, a, false, b, (short)0, c,
                                                 false, false);
}

// A-operand fragment (16x32 bf16). p = lane's row base (+k0 applied).
// Lane (hi, r): elements 0..7 -> K = e + 8*hi ; elements 8..15 -> K = 16 + e' + 8*hi.
static __device__ __forceinline__ v16bf load_a_frag(const __bf16* __restrict__ p, int hi) {
  const v8bf lo = *(const v8bf*)(p + 8 * hi);
  const v8bf hv = *(const v8bf*)(p + 16 + 8 * hi);
  v16bf a;
#pragma unroll
  for (int e = 0; e < 8; ++e) { a[e] = lo[e]; a[e + 8] = hv[e]; }
  return a;
}

// B-operand fragment (32x16 bf16) from W[N,K] row-major; elements 0..7 -> K = e + 8*hi ; elements 8..15 -> K = 16 + e' + 8*hi.
static __device__ __forceinline__ v16bf load_b_frag(const __bf16* __restrict__ p, int hi) {
  const v8bf lo = *(const v8bf*)(p + 8 * hi);
  const v8bf hv = *(const v8bf*)(p + 16 + 8 * hi);
  v16bf b;
#pragma unroll
  for (int e = 0; e < 8; ++e) { b[e] = lo[e]; b[e + 8] = hv[e]; }
  return b;
}

// ---------------------------------------------------------------------------
// TDM: DMA a [rows x 32] bf16 tile (row stride = row_stride_elems) into LDS,
// padded 16B after every 64B row -> LDS row pitch = 80B (40 elements).
// D# packing per CDNA5 ISA.
// ---------------------------------------------------------------------------
#ifdef HAVE_TDM
static __device__ __forceinline__ void tdm_load_tile(unsigned lds_off,
                                                     const void* gptr,
                                                     unsigned rows,
                                                     unsigned row_stride_elems) {
  const unsigned long long ga = (unsigned long long)gptr;
  u4i g0;
  g0[0] = 1u;                                                  // count=1
  g0[1] = lds_off;                                             // lds_addr
  g0[2] = (unsigned)ga;                                        // global_addr lo
  g0[3] = (unsigned)((ga >> 32) & 0x01FFFFFFu) | (2u << 30);   // addr hi | type=2
  v8i g1;
  // data_size=2B (code 1), pad_enable, pad_interval=16 DW (code 3), pad_amount=4 DW (code 3)
  g1[0] = (int)((1u << 16) | (1u << 20) | (3u << 22) | (3u << 25));
  g1[1] = (int)(32u << 16);                 // tensor_dim0[15:0] = 32
  g1[2] = (int)((rows & 0xFFFFu) << 16);    // tensor_dim0 hi=0 | tensor_dim1[15:0]
  g1[3] = (int)(32u << 16);                 // tensor_dim1 hi=0 | tile_dim0 = 32
  g1[4] = (int)(rows & 0xFFFFu);            // tile_dim1 = rows | tile_dim2 = 0
  g1[5] = (int)row_stride_elems;            // tensor_dim0_stride[31:0]
  g1[6] = 0;                                // stride hi | dim1_stride lo
  g1[7] = 0;
  const v4i z4 = {0, 0, 0, 0};
#if defined(__clang_major__) && __clang_major__ >= 23
  const v8i z8 = {0, 0, 0, 0, 0, 0, 0, 0};
  __builtin_amdgcn_tensor_load_to_lds(g0, g1, z4, z4, z8, 0);
#else
  __builtin_amdgcn_tensor_load_to_lds(g0, g1, z4, z4, 0);
#endif
}
#endif

// ---------------------------------------------------------------------------
// elementwise kernels
// ---------------------------------------------------------------------------
__global__ void cast_f32_to_bf16(const float* __restrict__ in,
                                 __bf16* __restrict__ out, long n) {
  long i = (long)blockIdx.x * blockDim.x + threadIdx.x;
  if (i < n) out[i] = (__bf16)in[i];
}

__global__ void rope_inplace_bf16(__bf16* __restrict__ t, long n_pairs, int S, int H) {
  long idx = (long)blockIdx.x * blockDim.x + threadIdx.x;
  if (idx >= n_pairs) return;
  const int i = (int)(idx % 64);
  long rest = idx / 64;
  const int h = (int)(rest % H);
  const long row = rest / H;  // b*S + s
  const int s = (int)(row % S);
  const float inv_freq = __expf(-0.14391156514261295f * (float)i);  // 10000^(-i/64)
  const float theta = (float)s * inv_freq;
  float sn, cs;
  sincosf(theta, &sn, &cs);
  const long off = row * (long)(H * 128) + (long)h * 128 + i;
  const float t0 = (float)t[off];
  const float t1 = (float)t[off + 64];
  t[off]      = (__bf16)(t0 * cs - t1 * sn);
  t[off + 64] = (__bf16)(t1 * cs + t0 * sn);
}

// v [B,S,H*128] -> vt [B*H, 128, S] (key dim contiguous)
__global__ void transpose_heads_bf16(const __bf16* __restrict__ v,
                                     __bf16* __restrict__ vt,
                                     long n, int S, int H) {
  long idx = (long)blockIdx.x * blockDim.x + threadIdx.x;
  if (idx >= n) return;
  const int D = H * 128;
  const int d = (int)(idx % D);
  const long row = idx / D;
  const int s = (int)(row % S);
  const long b = row / S;
  const int h = d >> 7;
  const int dd = d & 127;
  vt[(((long)b * H + h) * 128 + dd) * (long)S + s] = v[idx];
}

// ---------------------------------------------------------------------------
// shared GEMM epilogue
// ---------------------------------------------------------------------------
static __device__ __forceinline__ void gemm_store(v8f (&acc)[2][4], int m0, int n0,
                                                  int r, int hi,
                                                  float* __restrict__ Cf,
                                                  __bf16* __restrict__ Cb, int ldc,
                                                  const float* __restrict__ bias) {
#pragma unroll
  for (int mi = 0; mi < 2; ++mi) {
#pragma unroll
    for (int j = 0; j < 4; ++j) {
      const int nn = n0 + 16 * j + r;
      const float bv = bias ? bias[nn] : 0.0f;
#pragma unroll
      for (int v = 0; v < 8; ++v) {
        const long row = (long)m0 + mi * 16 + 8 * hi + v;
        const float val = acc[mi][j][v] + bv;
        const long idx = row * (long)ldc + nn;
        if (Cf) Cf[idx] = val;
        if (Cb) Cb[idx] = (__bf16)val;
      }
    }
  }
}

// ---------------------------------------------------------------------------
// NT GEMM, direct-global-load variant (fallback): C = A[M,K] @ W[N,K]^T
// block 256 thr (8 waves), block tile 64x256, wave tile 32x64; grid (N/256, M/64).
// ---------------------------------------------------------------------------
__global__ __launch_bounds__(256)
void gemm_nt_bf16(const __bf16* __restrict__ A, int lda,
                  const __bf16* __restrict__ W, int ldb,
                  float* __restrict__ Cf, __bf16* __restrict__ Cb, int ldc,
                  const float* __restrict__ bias, int K) {
  const int tid = threadIdx.x;
  const int wave = tid >> 5, lane = tid & 31;
  const int r = lane & 15, hi = lane >> 4;
  const int m0 = blockIdx.y * 64 + (wave >> 2) * 32;
  const int n0 = blockIdx.x * 256 + (wave & 3) * 64;

  v8f acc[2][4];
#pragma unroll
  for (int mi = 0; mi < 2; ++mi)
#pragma unroll
    for (int j = 0; j < 4; ++j) acc[mi][j] = zero8();

  const __bf16* a0p = A + (long)(m0 + r) * lda;
  const __bf16* a1p = A + (long)(m0 + 16 + r) * lda;
  const __bf16* bp[4];
#pragma unroll
  for (int j = 0; j < 4; ++j) bp[j] = W + (long)(n0 + 16 * j + r) * ldb;

  for (int k0 = 0; k0 < K; k0 += 32) {
    const v16bf a0 = load_a_frag(a0p + k0, hi);
    const v16bf a1 = load_a_frag(a1p + k0, hi);
#pragma unroll
    for (int j = 0; j < 4; ++j) {
      const v16bf b = load_b_frag(bp[j] + k0, hi);
      acc[0][j] = wmma_bf16(a0, b, acc[0][j]);
      acc[1][j] = wmma_bf16(a1, b, acc[1][j]);
    }
  }
  gemm_store(acc, m0, n0, r, hi, Cf, Cb, ldc, bias);
}

// ---------------------------------------------------------------------------
// NT GEMM, TDM-staged variant: A (64x32) and B (256x32) K-slabs double-buffered
// in LDS, DMA'd by tensor_load_to_lds issued from wave 0, TENSORcnt-synced.
// LDS row pitch = 40 elements (80B) via D# padding -> conflict-free frag reads.
// ---------------------------------------------------------------------------
#ifdef HAVE_TDM
__global__ __launch_bounds__(256)
void gemm_nt_bf16_tdm(const __bf16* __restrict__ A, int lda,
                      const __bf16* __restrict__ W, int ldb,
                      float* __restrict__ Cf, __bf16* __restrict__ Cb, int ldc,
                      const float* __restrict__ bias, int K) {
  constexpr int P = 40;  // padded LDS row pitch (elements)
  __shared__ __align__(16) __bf16 Abuf[2][64 * P];
  __shared__ __align__(16) __bf16 Bbuf[2][256 * P];

  const int tid = threadIdx.x;
  const int wave = tid >> 5, lane = tid & 31;
  const int r = lane & 15, hi = lane >> 4;
  const int mblk = blockIdx.y * 64;
  const int nblk = blockIdx.x * 256;
  const int mloc = (wave >> 2) * 32;   // wave's rows within A tile
  const int nloc = (wave & 3) * 64;    // wave's cols within B tile

  v8f acc[2][4];
#pragma unroll
  for (int mi = 0; mi < 2; ++mi)
#pragma unroll
    for (int j = 0; j < 4; ++j) acc[mi][j] = zero8();

  // prologue: fetch first K-slab
  if (wave == 0) {
    tdm_load_tile((unsigned)(unsigned long long)(const void*)&Abuf[0][0],
                  A + (long)mblk * lda, 64, (unsigned)lda);
    tdm_load_tile((unsigned)(unsigned long long)(const void*)&Bbuf[0][0],
                  W + (long)nblk * ldb, 256, (unsigned)ldb);
  }
  __builtin_amdgcn_s_wait_tensorcnt(0);
  __syncthreads();

  for (int k0 = 0; k0 < K; k0 += 32) {
    const int p = (k0 >> 5) & 1;
    // prefetch next slab into the other buffer while computing this one
    if (wave == 0 && (k0 + 32) < K) {
      tdm_load_tile((unsigned)(unsigned long long)(const void*)&Abuf[p ^ 1][0],
                    A + (long)mblk * lda + (k0 + 32), 64, (unsigned)lda);
      tdm_load_tile((unsigned)(unsigned long long)(const void*)&Bbuf[p ^ 1][0],
                    W + (long)nblk * ldb + (k0 + 32), 256, (unsigned)ldb);
    }

    const __bf16* a0p = &Abuf[p][(mloc + r) * P];
    const v16bf a0 = load_a_frag(a0p, hi);
    const v16bf a1 = load_a_frag(a0p + 16 * P, hi);
#pragma unroll
    for (int j = 0; j < 4; ++j) {
      const v16bf b = load_b_frag(&Bbuf[p][(nloc + 16 * j + r) * P], hi);
      acc[0][j] = wmma_bf16(a0, b, acc[0][j]);
      acc[1][j] = wmma_bf16(a1, b, acc[1][j]);
    }

    __builtin_amdgcn_s_wait_tensorcnt(0);  // wave0's DMA done (others: no-op)
    __syncthreads();                       // slab visible to all waves
  }
  gemm_store(acc, mblk + mloc, nblk + nloc, r, hi, Cf, Cb, ldc, bias);
}
#endif

// ---------------------------------------------------------------------------
// Flash attention (causal), bf16 WMMA, f32 online softmax.
// grid = (S/64, B*H), block = 128 (4 waves); each wave owns 16 query rows.
// ---------------------------------------------------------------------------
__global__ __launch_bounds__(128)
void mla_attn(const __bf16* __restrict__ q,
              const __bf16* __restrict__ k,
              const __bf16* __restrict__ vt,
              __bf16* __restrict__ ctx,
              int B, int S, int H) {
  const float scale = 0.08838834764831845f;  // 1/sqrt(128)
  const int tid = threadIdx.x;
  const int wave = tid >> 5, lane = tid & 31;
  const int r = lane & 15, hi = lane >> 4;
  const int qt = blockIdx.x;
  const int bh = blockIdx.y;
  const int b = bh / H, h = bh - b * H;
  const long D = (long)H * 128;
  const int qrow0 = qt * 64 + wave * 16;

  const __bf16* qb = q + ((long)b * S) * D + (long)h * 128;
  const __bf16* kb = k + ((long)b * S) * D + (long)h * 128;
  const __bf16* vb = vt + ((long)bh * 128) * (long)S;

  __shared__ __align__(16) __bf16 plds[4][16][64];

  v16bf qa[4];
  {
    const __bf16* qr = qb + (long)(qrow0 + r) * D;
#pragma unroll
    for (int f = 0; f < 4; ++f) qa[f] = load_a_frag(qr + 32 * f, hi);
  }

  v8f o[8];
#pragma unroll
  for (int j = 0; j < 8; ++j) o[j] = zero8();
  float mst[8], lst[8];
#pragma unroll
  for (int v = 0; v < 8; ++v) { mst[v] = -3.0e38f; lst[v] = 0.0f; }

  for (int kt = 0; kt <= qt; ++kt) {
    const int krow0 = kt * 64;

    // S = Q @ K^T  (16 WMMA)
    v8f sc[4];
#pragma unroll
    for (int j = 0; j < 4; ++j) sc[j] = zero8();
#pragma unroll
    for (int f = 0; f < 4; ++f) {
#pragma unroll
      for (int j = 0; j < 4; ++j) {
        const v16bf bf = load_b_frag(kb + (long)(krow0 + 16 * j + r) * D + 32 * f, hi);
        sc[j] = wmma_bf16(qa[f], bf, sc[j]);
      }
    }

    const bool diag = (kt == qt);
#pragma unroll
    for (int j = 0; j < 4; ++j) {
      const int col = krow0 + 16 * j + r;
#pragma unroll
      for (int v = 0; v < 8; ++v) {
        float val = sc[j][v] * scale;
        if (diag && col > qrow0 + 8 * hi + v) val = -3.0e38f;
        sc[j][v] = val;
      }
    }

    // online softmax (row m = v + 8*hi lives in the 16 lanes sharing hi)
#pragma unroll
    for (int v = 0; v < 8; ++v) {
      float mx = fmaxf(fmaxf(sc[0][v], sc[1][v]), fmaxf(sc[2][v], sc[3][v]));
#pragma unroll
      for (int d = 8; d >= 1; d >>= 1) mx = fmaxf(mx, __shfl_xor(mx, d, 32));
      const float mnew = fmaxf(mst[v], mx);
      const float alpha = __expf(mst[v] - mnew);
      float ps = 0.0f;
#pragma unroll
      for (int j = 0; j < 4; ++j) {
        const float pp = __expf(sc[j][v] - mnew);
        sc[j][v] = pp;
        ps += pp;
      }
#pragma unroll
      for (int d = 8; d >= 1; d >>= 1) ps += __shfl_xor(ps, d, 32);
      lst[v] = lst[v] * alpha + ps;
      mst[v] = mnew;
#pragma unroll
      for (int j = 0; j < 8; ++j) o[j][v] *= alpha;
    }

    // P: C-layout -> LDS -> A-layout bf16
#pragma unroll
    for (int j = 0; j < 4; ++j)
#pragma unroll
      for (int v = 0; v < 8; ++v)
        plds[wave][8 * hi + v][16 * j + r] = (__bf16)sc[j][v];
    __syncthreads();

    // O += P @ V  (16 WMMA)
#pragma unroll
    for (int t = 0; t < 2; ++t) {
      const v16bf pa = load_a_frag(&plds[wave][r][0] + 32 * t, hi);
#pragma unroll
      for (int j = 0; j < 8; ++j) {
        const v16bf bf = load_b_frag(vb + (long)(16 * j + r) * S + krow0 + 32 * t, hi);
        o[j] = wmma_bf16(pa, bf, o[j]);
      }
    }
    __syncthreads();
  }

#pragma unroll
  for (int v = 0; v < 8; ++v) {
    const float inv = 1.0f / lst[v];
    const long srow = qrow0 + 8 * hi + v;
    __bf16* crow = ctx + ((long)b * S + srow) * D + (long)h * 128;
#pragma unroll
    for (int j = 0; j < 8; ++j) crow[16 * j + r] = (__bf16)(o[j][v] * inv);
  }
}

// ---------------------------------------------------------------------------
// launcher
// ---------------------------------------------------------------------------
extern "C" void kernel_launch(void* const* d_in, const int* in_sizes, int n_in,
                              void* d_out, int out_size, void* d_ws, size_t ws_size,
                              hipStream_t stream) {
  (void)in_sizes; (void)n_in; (void)out_size; (void)ws_size;

  const int B = 2, S = 2048, D = 2048, L = 256, H = 16;
  const int M = B * S;  // 4096

  const float* x    = (const float*)d_in[0];
  const float* wq   = (const float*)d_in[1];
  const float* wdkv = (const float*)d_in[2];
  const float* wuk  = (const float*)d_in[3];
  const float* wuv  = (const float*)d_in[4];
  const float* wo   = (const float*)d_in[5];
  const float* bo   = (const float*)d_in[6];
  float* out = (float*)d_out;

  char* ws = (char*)d_ws;
  size_t off = 0;
  auto alloc = [&](size_t bytes) -> void* {
    void* p = ws + off;
    off += (bytes + 255) & ~(size_t)255;
    return p;
  };
  __bf16* xb    = (__bf16*)alloc((size_t)M * D * 2);
  __bf16* wqb   = (__bf16*)alloc((size_t)D * D * 2);
  __bf16* wdkvb = (__bf16*)alloc((size_t)L * D * 2);
  __bf16* wukb  = (__bf16*)alloc((size_t)D * L * 2);
  __bf16* wuvb  = (__bf16*)alloc((size_t)D * L * 2);
  __bf16* wob   = (__bf16*)alloc((size_t)D * D * 2);
  __bf16* qbuf  = (__bf16*)alloc((size_t)M * D * 2);
  __bf16* latb  = (__bf16*)alloc((size_t)M * L * 2);
  __bf16* kbuf  = (__bf16*)alloc((size_t)M * D * 2);
  __bf16* vbuf  = (__bf16*)alloc((size_t)M * D * 2);
  __bf16* vtb   = (__bf16*)alloc((size_t)M * D * 2);
  __bf16* ctxb  = (__bf16*)alloc((size_t)M * D * 2);

  auto cast = [&](const float* src, __bf16* dst, long n) {
    cast_f32_to_bf16<<<(unsigned)((n + 255) / 256), 256, 0, stream>>>(src, dst, n);
  };
  cast(x,    xb,    (long)M * D);
  cast(wq,   wqb,   (long)D * D);
  cast(wdkv, wdkvb, (long)L * D);
  cast(wuk,  wukb,  (long)D * L);
  cast(wuv,  wuvb,  (long)D * L);
  cast(wo,   wob,   (long)D * D);

  const dim3 blk256(256), blk128(128);
  const dim3 gFull(D / 256, M / 64);              // (8, 64)
  const dim3 gLat(1, M / 64);                     // N = 256 -> one column of blocks

#ifdef HAVE_TDM
#define GEMM gemm_nt_bf16_tdm
#else
#define GEMM gemm_nt_bf16
#endif

  // q = x @ wq^T
  GEMM<<<gFull, blk256, 0, stream>>>(xb, D, wqb, D, nullptr, qbuf, D, nullptr, D);
  // latent = x @ wdkv^T
  GEMM<<<gLat, blk256, 0, stream>>>(xb, D, wdkvb, D, nullptr, latb, L, nullptr, D);
  // k = latent @ wuk^T
  GEMM<<<gFull, blk256, 0, stream>>>(latb, L, wukb, L, nullptr, kbuf, D, nullptr, L);
  // v = latent @ wuv^T
  GEMM<<<gFull, blk256, 0, stream>>>(latb, L, wuvb, L, nullptr, vbuf, D, nullptr, L);

  // RoPE in place on q, k
  const long n_pairs = (long)M * H * 64;
  rope_inplace_bf16<<<(unsigned)((n_pairs + 255) / 256), blk256, 0, stream>>>(qbuf, n_pairs, S, H);
  rope_inplace_bf16<<<(unsigned)((n_pairs + 255) / 256), blk256, 0, stream>>>(kbuf, n_pairs, S, H);

  // v -> vt [B*H, 128, S]
  const long nv = (long)M * D;
  transpose_heads_bf16<<<(unsigned)((nv + 255) / 256), blk256, 0, stream>>>(vbuf, vtb, nv, S, H);

  // flash attention
  mla_attn<<<dim3(S / 64, B * H), blk128, 0, stream>>>(qbuf, kbuf, vtb, ctxb, B, S, H);

  // out = ctx @ wo^T + bo  (f32 into d_out)
  GEMM<<<gFull, blk256, 0, stream>>>(ctxb, D, wob, D, out, nullptr, D, bo, D);
}
